// PDN_dense_add_skip_4bro_pre_batch_act_max_test_38010460569944
// MI455X (gfx1250) — matrix operand order repaired and checked by reading, not compiled, not run on hardware
//
#include <hip/hip_runtime.h>

#define N_NODES 50000
#define N_EDGES 800000
#define F_IN    128
#define DMODEL  256
#define EDIM    16
#define NCLS    2
#define NGRAPH  500
#define BN_EPS  1e-5f

typedef float v8f __attribute__((ext_vector_type(8)));
typedef __attribute__((ext_vector_type(16))) __bf16 v16bf;

__device__ __forceinline__ unsigned short f2bf(float f) {
    unsigned int u = __float_as_uint(f);
    u += 0x7fffu + ((u >> 16) & 1u);            // round-to-nearest-even
    return (unsigned short)(u >> 16);
}

// ---------------- elementwise / small kernels ----------------

__global__ void cvt_f32_to_bf16_kernel(const float* __restrict__ x,
                                       unsigned short* __restrict__ y, int n) {
    int t = blockIdx.x * blockDim.x + threadIdx.x;
    if (t < n) y[t] = f2bf(x[t]);
}

// W [K, 256] row-major -> Wt [256, K] bf16 (column-major pack for B fragments)
__global__ void pack_w_kernel(const float* __restrict__ W,
                              unsigned short* __restrict__ Wt, int K) {
    int t = blockIdx.x * blockDim.x + threadIdx.x;
    if (t >= K * DMODEL) return;
    int k = t / DMODEL;
    int n = t - k * DMODEL;
    Wt[(size_t)n * K + k] = f2bf(W[t]);
}

// per-edge MLP: w = sigmoid(relu(ea@We1 + be1) @ We2 + be2)
__global__ void edge_mlp_kernel(const float* __restrict__ ea,
                                const float* __restrict__ We1,
                                const float* __restrict__ be1,
                                const float* __restrict__ We2,
                                const float* __restrict__ be2,
                                float* __restrict__ w, int E) {
    int e = blockIdx.x * blockDim.x + threadIdx.x;
    if (e >= E) return;
    float a[16];
    const float4* p = (const float4*)(ea + (size_t)e * EDIM);
    *(float4*)&a[0]  = p[0];
    *(float4*)&a[4]  = p[1];
    *(float4*)&a[8]  = p[2];
    *(float4*)&a[12] = p[3];
    float z = be2[0];
#pragma unroll
    for (int j = 0; j < 16; ++j) {
        float h = be1[j];
#pragma unroll
        for (int i = 0; i < 16; ++i) h = fmaf(a[i], We1[i * 16 + j], h);
        h = fmaxf(h, 0.f);
        z = fmaf(h, We2[j], z);
    }
    w[e] = 1.f / (1.f + __expf(-z));
}

__global__ void deg_init_kernel(float* __restrict__ deg, int n) {
    int t = blockIdx.x * blockDim.x + threadIdx.x;
    if (t < n) deg[t] = 1.0f;                    // self-loop weight
}

__global__ void deg_acc_kernel(const int* __restrict__ dst,
                               const float* __restrict__ w,
                               float* __restrict__ deg, int E) {
    int e = blockIdx.x * blockDim.x + threadIdx.x;
    if (e < E) atomicAdd(&deg[dst[e]], w[e]);
}

__global__ void dinv_kernel(const float* __restrict__ deg,
                            float* __restrict__ dinv, int n) {
    int t = blockIdx.x * blockDim.x + threadIdx.x;
    if (t < n) {
        float d = deg[t];
        dinv[t] = d > 0.f ? rsqrtf(d) : 0.f;
    }
}

// in-place: w[e] <- dinv[src]*w[e]*dinv[dst]
__global__ void gcn_norm_kernel(const int* __restrict__ src,
                                const int* __restrict__ dst,
                                const float* __restrict__ dinv,
                                float* __restrict__ w, int E) {
    int e = blockIdx.x * blockDim.x + threadIdx.x;
    if (e < E) w[e] = dinv[src[e]] * w[e] * dinv[dst[e]];
}

// out[i,d] = dinv[i]^2 * xt[i,d] + bias[d]   (self-loop term + bias, also zero-base for atomics)
__global__ void prefill_kernel(const float* __restrict__ xt,
                               const float* __restrict__ dinv,
                               const float* __restrict__ bias,
                               float* __restrict__ out) {
    int t = blockIdx.x * blockDim.x + threadIdx.x;
    if (t >= N_NODES * DMODEL) return;
    int i = t >> 8;
    int d = t & 255;
    float di = dinv[i];
    out[t] = fmaf(di * di, xt[t], bias[d]);
}

// one wave per edge: out[dst] += c * xt[src]
__global__ __launch_bounds__(256)
void edge_agg_kernel(const int* __restrict__ src, const int* __restrict__ dst,
                     const float* __restrict__ nw, const float* __restrict__ xt,
                     float* __restrict__ out, int E) {
    int wave = (int)((blockIdx.x * blockDim.x + threadIdx.x) >> 5);
    int lane = threadIdx.x & 31;
    if (wave >= E) return;
    int s = src[wave];
    int d = dst[wave];
    float c = nw[wave];
    const float4* xp = (const float4*)(xt + (size_t)s * DMODEL + lane * 8);
    float4 a = xp[0], b = xp[1];
    float* op = out + (size_t)d * DMODEL + lane * 8;
    atomicAdd(op + 0, c * a.x);
    atomicAdd(op + 1, c * a.y);
    atomicAdd(op + 2, c * a.z);
    atomicAdd(op + 3, c * a.w);
    atomicAdd(op + 4, c * b.x);
    atomicAdd(op + 5, c * b.y);
    atomicAdd(op + 6, c * b.z);
    atomicAdd(op + 7, c * b.w);
}

__global__ void zero2_kernel(float* __restrict__ a, float* __restrict__ b, int n) {
    int t = blockIdx.x * blockDim.x + threadIdx.x;
    if (t < n) { a[t] = 0.f; b[t] = 0.f; }
}

// one column per thread, coalesced row reads, local accumulation then atomics
__global__ __launch_bounds__(256)
void bn_stats_kernel(const float* __restrict__ x,
                     float* __restrict__ sums, float* __restrict__ sqs) {
    int d = threadIdx.x;
    float s = 0.f, q = 0.f;
    for (int i = blockIdx.x; i < N_NODES; i += gridDim.x) {
        float v = x[(size_t)i * DMODEL + d];
        s += v;
        q = fmaf(v, v, q);
    }
    atomicAdd(&sums[d], s);
    atomicAdd(&sqs[d], q);
}

// y_bf16 = bf16(relu((x - mu) * rsqrt(var + eps)))
__global__ void bn_apply_kernel(const float* __restrict__ x,
                                const float* __restrict__ sums,
                                const float* __restrict__ sqs,
                                unsigned short* __restrict__ y) {
    int t = blockIdx.x * blockDim.x + threadIdx.x;
    if (t >= N_NODES * DMODEL) return;
    int d = t & 255;
    const float inv_n = 1.0f / (float)N_NODES;
    float mu = sums[d] * inv_n;
    float var = fmaf(sqs[d], inv_n, -mu * mu);
    float istd = rsqrtf(var + BN_EPS);
    float v = fmaxf((x[t] - mu) * istd, 0.f);
    y[t] = f2bf(v);
}

__global__ void add_inplace_kernel(float* __restrict__ a,
                                   const float* __restrict__ b, int n) {
    int t = blockIdx.x * blockDim.x + threadIdx.x;
    if (t < n) a[t] += b[t];
}

__global__ void copy_kernel(float* __restrict__ a, const float* __restrict__ b, int n) {
    int t = blockIdx.x * blockDim.x + threadIdx.x;
    if (t < n) a[t] = b[t];
}

__global__ void pool_init_kernel(float* __restrict__ pool, int n) {
    int t = blockIdx.x * blockDim.x + threadIdx.x;
    if (t < n) ((int*)pool)[t] = 0xFF800000;     // -inf bit pattern
}

// segment_max of relu(x): non-negative floats -> integer atomicMax on bits is order-preserving
__global__ void pool_max_kernel(const float* __restrict__ x,
                                const int* __restrict__ batch,
                                float* __restrict__ pool) {
    int t = blockIdx.x * blockDim.x + threadIdx.x;
    if (t >= N_NODES * DMODEL) return;
    int i = t >> 8;
    int d = t & 255;
    float v = fmaxf(x[t], 0.f);
    atomicMax((int*)&pool[(size_t)batch[i] * DMODEL + d], __float_as_int(v));
}

__global__ void classifier_kernel(const float* __restrict__ pool,
                                  const float* __restrict__ lw,
                                  const float* __restrict__ lb,
                                  float* __restrict__ out) {
    int t = blockIdx.x * blockDim.x + threadIdx.x;
    if (t >= NGRAPH * NCLS) return;
    int g = t / NCLS;
    int c = t - g * NCLS;
    float acc = lb[c];
    const float* pr = pool + (size_t)g * DMODEL;
    for (int d = 0; d < DMODEL; ++d) acc = fmaf(pr[d], lw[d * NCLS + c], acc);
    out[t] = acc;
}

// ---------------- WMMA bf16 GEMM:  XT[M,256] = Xbf[M,K] @ W[K,256] ----------------
// Wt is W transposed & packed bf16: Wt[n*K + k].
// One wave per 16-row strip; 16 resident f32 accumulators (full 256-col strip).
__global__ __launch_bounds__(256)
void gemm_bf16_kernel(const unsigned short* __restrict__ Xbf,
                      const unsigned short* __restrict__ Wt,
                      float* __restrict__ XT, int M, int K) {
    int wave = threadIdx.x >> 5;
    int lane = threadIdx.x & 31;
    int gw = blockIdx.x * 8 + wave;              // global wave = M-tile index
    int mtiles = M >> 4;
    if (gw >= mtiles) return;                    // wave-uniform: EXEC stays all-ones
    int m0 = gw << 4;
    int l  = lane & 15;
    int hi = lane >> 4;

    v8f acc[16] = {};

    // A fragment (16x32 bf16): lane<16 holds row m0+l, K {k0..k0+7, k0+16..k0+23};
    // lane>=16 holds row m0+l, K {k0+8..k0+15, k0+24..k0+31}.
    const unsigned short* arow = Xbf + (size_t)(m0 + l) * K + hi * 8;

    for (int k0 = 0; k0 < K; k0 += 32) {
        union { v16bf v; uint4 u[2]; } A;
        const uint4* ap = (const uint4*)(arow + k0);
        A.u[0] = ap[0];
        A.u[1] = ap[2];                          // +16 bf16 elements
        __builtin_prefetch(arow + k0 + 32, 0, 0);
#pragma unroll
        for (int t = 0; t < 16; ++t) {
            // B fragment (32x16 bf16): lane<16 holds col n0+l K {k0..k0+7, k0+16..k0+23};
            // lane>=16 holds col n0+l K {k0+8..k0+15, k0+24..k0+31}. Wt row is contiguous in K.
            union { v16bf v; uint4 u[2]; } B;
            const uint4* bp =
                (const uint4*)(Wt + (size_t)(t * 16 + l) * K + k0 + hi * 8);
            B.u[0] = bp[0];
            B.u[1] = bp[2];
            acc[t] = __builtin_amdgcn_wmma_f32_16x16x32_bf16(
                false, A.v, false, B.v, (short)0, acc[t], false, false);
        }
    }

    // C/D layout: VGPR r -> (m0 + r + hi*8, n0 + l)
#pragma unroll
    for (int t = 0; t < 16; ++t) {
        float* orow = XT + (size_t)(m0 + hi * 8) * DMODEL + t * 16 + l;
#pragma unroll
        for (int r = 0; r < 8; ++r) orow[(size_t)r * DMODEL] = acc[t][r];
    }
}

// ---------------- host orchestration ----------------

static inline int cdiv(long long a, int b) { return (int)((a + b - 1) / b); }

extern "C" void kernel_launch(void* const* d_in, const int* in_sizes, int n_in,
                              void* d_out, int out_size, void* d_ws, size_t ws_size,
                              hipStream_t stream) {
    const float* X0    = (const float*)d_in[0];
    const int*   EIDX  = (const int*)d_in[1];
    const int*   SRC   = EIDX;
    const int*   DST   = EIDX + N_EDGES;
    const int*   BATCH = (const int*)d_in[2];
    const float* EATTR = (const float*)d_in[4];
    const float* W_in  = (const float*)d_in[5];
    const float* b_in  = (const float*)d_in[6];
    const float* We1_in = (const float*)d_in[7];
    const float* be1_in = (const float*)d_in[8];
    const float* We2_in = (const float*)d_in[9];
    const float* be2_in = (const float*)d_in[10];
    const float* Wh    = (const float*)d_in[11];
    const float* bh    = (const float*)d_in[12];
    const float* We1h  = (const float*)d_in[13];
    const float* be1h  = (const float*)d_in[14];
    const float* We2h  = (const float*)d_in[15];
    const float* be2h  = (const float*)d_in[16];
    const float* linW  = (const float*)d_in[17];
    const float* linb  = (const float*)d_in[18];
    float* OUT = (float*)d_out;

    // workspace carve (256B aligned)
    char* wsp = (char*)d_ws;
    auto carve = [&](size_t bytes) -> void* {
        void* p = (void*)wsp;
        wsp += (bytes + 255) & ~(size_t)255;
        return p;
    };
    const size_t ND = (size_t)N_NODES * DMODEL;
    float* XA   = (float*)carve(ND * 4);
    float* XB   = (float*)carve(ND * 4);
    float* SSUM = (float*)carve(ND * 4);
    float* XT   = (float*)carve(ND * 4);
    unsigned short* XBF = (unsigned short*)carve(ND * 2);
    unsigned short* WT  = (unsigned short*)carve((size_t)DMODEL * DMODEL * 2);
    float* EW   = (float*)carve((size_t)N_EDGES * 4);
    float* DEG  = (float*)carve((size_t)N_NODES * 4);
    float* DINV = (float*)carve((size_t)N_NODES * 4);
    float* SUMS = (float*)carve(DMODEL * 4);
    float* SQS  = (float*)carve(DMODEL * 4);
    float* POOL = (float*)carve((size_t)NGRAPH * DMODEL * 4);

    const int TB = 256;
    const int mtiles = N_NODES / 16;             // 3125 exactly
    const int gemm_blocks = cdiv(mtiles, 8);

    auto run_conv = [&](const unsigned short* xbf, int K,
                        const float* W, const float* b,
                        const float* We1, const float* be1,
                        const float* We2, const float* be2,
                        float* out_x) {
        pack_w_kernel<<<cdiv((long long)K * DMODEL, TB), TB, 0, stream>>>(W, WT, K);
        edge_mlp_kernel<<<cdiv(N_EDGES, TB), TB, 0, stream>>>(EATTR, We1, be1, We2, be2, EW, N_EDGES);
        deg_init_kernel<<<cdiv(N_NODES, TB), TB, 0, stream>>>(DEG, N_NODES);
        deg_acc_kernel<<<cdiv(N_EDGES, TB), TB, 0, stream>>>(DST, EW, DEG, N_EDGES);
        dinv_kernel<<<cdiv(N_NODES, TB), TB, 0, stream>>>(DEG, DINV, N_NODES);
        gcn_norm_kernel<<<cdiv(N_EDGES, TB), TB, 0, stream>>>(SRC, DST, DINV, EW, N_EDGES);
        gemm_bf16_kernel<<<gemm_blocks, TB, 0, stream>>>(xbf, WT, XT, N_NODES, K);
        prefill_kernel<<<cdiv((long long)ND, TB), TB, 0, stream>>>(XT, DINV, b, out_x);
        edge_agg_kernel<<<cdiv((long long)N_EDGES * 32, TB), TB, 0, stream>>>(SRC, DST, EW, XT, out_x, N_EDGES);
    };

    // conv1: x -> XA (input converted to bf16, K = 128)
    cvt_f32_to_bf16_kernel<<<cdiv((long long)N_NODES * F_IN, TB), TB, 0, stream>>>(X0, XBF, N_NODES * F_IN);
    run_conv(XBF, F_IN, W_in, b_in, We1_in, be1_in, We2_in, be2_in, XA);

    // running skip sum S = x0
    copy_kernel<<<cdiv((long long)ND, TB), TB, 0, stream>>>(SSUM, XA, (int)ND);

    float* Xc = XA;
    float* Xo = XB;
    for (int g = 0; g < 4; ++g) {
        for (int i = 0; i < 2; ++i) {
            zero2_kernel<<<1, 256, 0, stream>>>(SUMS, SQS, DMODEL);
            bn_stats_kernel<<<256, 256, 0, stream>>>(Xc, SUMS, SQS);
            bn_apply_kernel<<<cdiv((long long)ND, TB), TB, 0, stream>>>(Xc, SUMS, SQS, XBF);
            int li = g * 2 + i;
            run_conv(XBF, DMODEL,
                     Wh + (size_t)li * DMODEL * DMODEL, bh + (size_t)li * DMODEL,
                     We1h + (size_t)li * EDIM * EDIM, be1h + (size_t)li * EDIM,
                     We2h + (size_t)li * EDIM, be2h + (size_t)li,
                     Xo);
            float* t = Xc; Xc = Xo; Xo = t;
        }
        // x += S; S += x   (== dense additive skips + append)
        add_inplace_kernel<<<cdiv((long long)ND, TB), TB, 0, stream>>>(Xc, SSUM, (int)ND);
        add_inplace_kernel<<<cdiv((long long)ND, TB), TB, 0, stream>>>(SSUM, Xc, (int)ND);
    }

    // relu + global max pool + classifier
    pool_init_kernel<<<cdiv((long long)NGRAPH * DMODEL, TB), TB, 0, stream>>>(POOL, NGRAPH * DMODEL);
    pool_max_kernel<<<cdiv((long long)ND, TB), TB, 0, stream>>>(Xc, BATCH, POOL);
    classifier_kernel<<<cdiv(NGRAPH * NCLS, TB), TB, 0, stream>>>(POOL, linW, linb, OUT);

    (void)in_sizes; (void)n_in; (void)out_size; (void)ws_size;
}
